// CTLSTM_59742995088104
// MI455X (gfx1250) — matrix operand order repaired and checked by reading, not compiled, not run on hardware
//
#include <hip/hip_runtime.h>

// Time-aware conv-LSTM (TLSTMv2) for MI455X / gfx1250.
// Convs expressed as im2col GEMMs on V_WMMA_F32_16X16X32_F16 (K=25 padded to 32).
// One wave32 per batch element.
//
// Round-3 changes:
//  - LDS holds the im2col matrices directly, column-major im[col][kk]: staging
//    scatters each value to its 5 (col,kk) slots; WMMA B operands load as two
//    ds_load_b128 per fragment (was 16x ds_load_u16 + v_mov_b16 packing).
//  - Gate/decay biases folded into the GEMM via the spare kk=25 column
//    (A[m][25]=bias[m], im_x/im_c keep a permanent 1.0 in row kk=25).
//  - f32 cell state kept in 10 registers per lane (no LDS round trip).

#define T_STEPS 8
#define NBATCH  16384
#define C_CH    5
#define L_LEN   64
#define KKDIM   25                 // C_CH * 5 taps
#define IM_COLS (L_LEN + 4)        // logical col l lives at row l+2; rows 0,1,66,67 = dump margin
#define IM_K    40                 // halves per row: kk 0..24 data, 25 = bias-one, 26..39 zero pad
#define IM_SZ   (IM_COLS * IM_K)   // 2720 halves = 5440 B (80 B row keeps 16 B align)
#define WAVES   2
#define THREADS (WAVES * 32)
#define E_CONST 2.7182818284590452f

typedef __attribute__((ext_vector_type(16))) _Float16 v16h;
typedef __attribute__((ext_vector_type(8)))  _Float16 v8h;
typedef __attribute__((ext_vector_type(8)))  float    v8f;

__device__ __forceinline__ void lds_fence() {
  // per-wave LDS ops are in-order; this blocks compiler reordering and drains DScnt
  asm volatile("s_wait_dscnt 0" ::: "memory");
}

__device__ __forceinline__ float fast_rcp(float x) {
  return __builtin_amdgcn_rcpf(x);
}

#if __has_builtin(__builtin_amdgcn_tanhf)
__device__ __forceinline__ float fast_tanh(float x) {
  return __builtin_amdgcn_tanhf(x);
}
#elif __has_builtin(__builtin_amdgcn_tanh_f32)
__device__ __forceinline__ float fast_tanh(float x) {
  return __builtin_amdgcn_tanh_f32(x);
}
#else
__device__ __forceinline__ float fast_tanh(float x) {
  float e = __expf(-2.0f * x);
  return (1.0f - e) * fast_rcp(1.0f + e);
}
#endif

__device__ __forceinline__ float fast_sigmoid(float x) {
  return __builtin_fmaf(0.5f, fast_tanh(0.5f * x), 0.5f);
}

__device__ __forceinline__ v8f wmma_f16(v16h a, v16h b, v8f c) {
  // (neg_a, A, neg_b, B, c_mod, C, reuse_a, reuse_b)
  return __builtin_amdgcn_wmma_f32_16x16x32_f16(false, a, false, b, (short)0, c,
                                                false, false);
}

// A operand (M=16 x K=32, f16). ISA layout (CDNA5 ISA, 16-bit A 16x32):
//   lanes 0-15 : V0..3 -> K=0..7,  V4..7 -> K=16..23   (M = lane)
//   lanes 16-31: V0..3 -> K=8..15, V4..7 -> K=24..31   (M = lane-16)
// Rows [baseO, maxO) from W[o][i][k] (flat o*25 + kk); kk=25 carries bias[o]
// (paired with the permanent 1.0 in the B operand's kk=25 row); rest zero.
__device__ __forceinline__ v16h load_A(const float* __restrict__ W,
                                       const float* __restrict__ bias,
                                       int baseO, int maxO, int lane) {
  int m = lane & 15;
  int half = lane >> 4;
  v16h a = {};
  int o = baseO + m;
  if (o < maxO) {
#pragma unroll
    for (int j = 0; j < 8; ++j) {
      int ke = 2 * j + (j < 4 ? (half ? 8 : 0) : (half ? 16 : 8));
      float v0 = (ke < KKDIM) ? W[o * KKDIM + ke] : 0.0f;
      float v1 = (ke + 1 < KKDIM) ? W[o * KKDIM + ke + 1] : 0.0f;
      a[2 * j]     = (_Float16)v0;
      a[2 * j + 1] = (_Float16)v1;
    }
    if (half) a[9] = (_Float16)bias[o];  // element 9 of upper lanes == kk 25
  }
  return a;
}

// B operand (K=32 x N=16, f16): lanes 0-15 hold K=0..7,16..23 (col = lane),
// lanes 16-31 hold K=8..15,24..31 (col = lane-16). With column-major im2col in LDS
// this is two 8-half chunks -> two ds_load_b128.
__device__ __forceinline__ v16h build_B(const _Float16* __restrict__ im,
                                        int colBase, int lane) {
  const int row = colBase + (lane & 15) + 2;  // +2 dump-margin offset
  const int kk0 = (lane & 16) >> 1;
  const v8h* p = (const v8h*)(im + row * IM_K + kk0);
  v8h lo = p[0];
  v8h hi = p[2];
  return __builtin_shufflevector(lo, hi, 0, 1, 2, 3, 4, 5, 6, 7,
                                 8, 9, 10, 11, 12, 13, 14, 15);
}

// Scatter one signal value to its 5 im2col slots:
// im[(p+2) - (k-2)][ch*5 + k] = v  for k = 0..4 (margins absorb the edges).
__device__ __forceinline__ void scatter_im(_Float16* __restrict__ im, int ch,
                                           int p, float v) {
  _Float16 hv = (_Float16)v;
#pragma unroll
  for (int k = 0; k < 5; ++k)
    im[(p + 4 - k) * IM_K + ch * 5 + k] = hv;
}

__global__ __launch_bounds__(THREADS) void ctlstm_wmma_kernel(
    const float* __restrict__ x,      // (T, B, C, L)
    const float* __restrict__ td,     // (B, T)
    const float* __restrict__ W_ih,   // (20, 5, 5)
    const float* __restrict__ W_hh,   // (20, 5, 5)
    const float* __restrict__ bias,   // (20,)
    const float* __restrict__ W_d,    // (5, 5, 5)
    const float* __restrict__ b_d,    // (5,)
    const float* __restrict__ fc2w,   // (64,)
    const float* __restrict__ fc2b,   // (1,)
    const float* __restrict__ basep,  // (1,)
    float* __restrict__ out)          // (B,)
{
  __shared__ __align__(16) _Float16 s_imx[WAVES][IM_SZ];
  __shared__ __align__(16) _Float16 s_imh[WAVES][IM_SZ];
  __shared__ __align__(16) _Float16 s_imc[WAVES][IM_SZ];
  __shared__ float s_cs[WAVES][C_CH * L_LEN];
  __shared__ float s_g [WAVES][4 * C_CH * L_LEN];
  __shared__ float cb[66];  // [0..63] fc2_w, [64] fc2_b, [65] baseline

  const int tid  = threadIdx.x;
  const int lane = tid & 31;
  const int wv   = tid >> 5;
  const int b    = blockIdx.x * WAVES + wv;

  // block-shared head constants
  for (int i = tid; i < L_LEN; i += THREADS) cb[i] = fc2w[i];
  if (tid == 0) { cb[64] = fc2b[0]; cb[65] = basep[0]; }

  _Float16* imx = s_imx[wv];
  _Float16* imh = s_imh[wv];
  _Float16* imc = s_imc[wv];
  float*    scs = s_cs[wv];
  float*    sg  = s_g[wv];

  // zero im2col blocks (kk pad + out-of-range taps stay zero forever), then plant the
  // bias-one column (kk=25) in the x and c blocks.
  for (int i = lane; i < IM_SZ; i += 32) {
    imx[i] = (_Float16)0.0f; imh[i] = (_Float16)0.0f; imc[i] = (_Float16)0.0f;
  }
  for (int r = lane; r < IM_COLS; r += 32) {
    imx[r * IM_K + KKDIM] = (_Float16)1.0f;
    imc[r * IM_K + KKDIM] = (_Float16)1.0f;
  }

  __syncthreads();
  if (b >= NBATCH) return;  // wave-uniform (keeps EXEC all-ones at WMMA sites)

  // weight fragments (once per wave; tiny, L2-resident). Biases ride in kk=25.
  const v16h a_ih0 = load_A(W_ih, bias, 0, 20, lane);
  const v16h a_ih1 = load_A(W_ih, bias, 16, 20, lane);
  const v16h a_hh0 = load_A(W_hh, bias, 0, 20, lane);   // pairs with imh kk25==0
  const v16h a_hh1 = load_A(W_hh, bias, 16, 20, lane);  // -> bias added once
  const v16h a_d   = load_A(W_d, b_d, 0, C_CH, lane);

  const int col16 = lane & 15;
  const int rb    = (lane >> 4) * 8;

  float creg[10];  // f32 cell state, lane-private: creg[it] = c[(it*32+lane)]
#pragma unroll
  for (int it = 0; it < 10; ++it) creg[it] = 0.0f;

  for (int t = 0; t < T_STEPS; ++t) {
    // ---- stage x_t into its im2col block; prefetch x_{t+1}
    const float* xp = x + ((size_t)t * NBATCH + b) * (C_CH * L_LEN);
#pragma unroll
    for (int it = 0; it < 10; ++it) {
      int ii = it * 32 + lane;
      scatter_im(imx, ii >> 6, ii & 63, xp[ii]);
    }
    if (t + 1 < T_STEPS) {
      const float* xn = xp + (size_t)NBATCH * C_CH * L_LEN;
#pragma unroll
      for (int base = 0; base < C_CH * L_LEN; base += 32)
        __builtin_prefetch(xn + base + lane, 0, 1);
    }

    float t2v = td[(size_t)b * T_STEPS + t];
    float t1v = (t == 0) ? t2v : td[(size_t)b * T_STEPS + t - 1];
    float d1 = fast_rcp(__logf(E_CONST + t1v));
    float d2 = fast_rcp(__logf(E_CONST + t2v));

    lds_fence();  // x im2col visible; prior-step h/c im2col visible

    // ---- WMMA: short-term memory conv (W_d*c + b_d) and gates (W_ih*x + W_hh*h + b)
#pragma unroll
    for (int ct = 0; ct < 4; ++ct) {
      const int col = ct * 16 + col16;

      // cs tile: rows 0..4 valid; they live in lanes 0..15's VGPRs 0..4
      v16h bc = build_B(imc, ct * 16, lane);
      v8f dcs = {};
      dcs = wmma_f16(a_d, bc, dcs);
      if (lane < 16) {
#pragma unroll
        for (int r = 0; r < C_CH; ++r)
          scs[r * L_LEN + col] = fast_tanh(dcs[r]);   // b_d folded via kk25
      }

      // gate tiles: 20 output channels -> row-tile0 (o 0..15) + row-tile1 (o 16..19)
      v16h bx = build_B(imx, ct * 16, lane);
      v16h bh = build_B(imh, ct * 16, lane);
      v8f g0 = {};
      g0 = wmma_f16(a_ih0, bx, g0);
      g0 = wmma_f16(a_hh0, bh, g0);
      v8f g1 = {};
      g1 = wmma_f16(a_ih1, bx, g1);
      g1 = wmma_f16(a_hh1, bh, g1);
#pragma unroll
      for (int r = 0; r < 8; ++r) sg[(rb + r) * L_LEN + col] = g0[r];
      if (lane < 16) {
#pragma unroll
        for (int r = 0; r < 4; ++r) sg[(16 + r) * L_LEN + col] = g1[r];
      }
    }

    lds_fence();  // gates + cs visible to elementwise phase

    // ---- elementwise TLSTM update (c state in registers; f16 im2col mirrors)
#pragma unroll
    for (int it = 0; it < 10; ++it) {
      int ii = it * 32 + lane;       // == ch*64 + p
      int ch = ii >> 6, p = ii & 63;
      float c_old = creg[it];
      float csv   = scs[ii];
      float c_adj = c_old - csv + csv * d1;
      float giv = sg[ii];            // biases already folded into the GEMM
      float gfv = sg[ii + 320];
      float ggv = sg[ii + 640];
      float gov = sg[ii + 960];
      float fg  = fast_sigmoid(gfv) * d2;
      float cn  = fg * c_adj + fast_sigmoid(giv) * fast_tanh(ggv);
      float hn  = fast_sigmoid(gov) * fast_tanh(cn);
      creg[it] = cn;
      scatter_im(imc, ch, p, cn);
      scatter_im(imh, ch, p, hn);
    }
    // next iteration's lds_fence orders these stores before the WMMA B-loads
  }

  lds_fence();

  // ---- head: out2[c] = sigmoid(h[c,:].fc2_w + fc2_b); case_pred from product.
  // h[ch][l] is readable at im_h[(l+2)][ch*5+2] (center-tap column).
  float prod = 1.0f;
#pragma unroll
  for (int ch = 0; ch < C_CH; ++ch) {
    float v = (float)imh[(lane + 2) * IM_K + ch * 5 + 2] * cb[lane] +
              (float)imh[(lane + 34) * IM_K + ch * 5 + 2] * cb[lane + 32];
#pragma unroll
    for (int off = 16; off > 0; off >>= 1) v += __shfl_xor(v, off, 32);
    prod *= (1.0f - fast_sigmoid(v + cb[64]));
  }
  if (lane == 0) out[b] = 1.0f - prod * (1.0f - fast_sigmoid(cb[65]));
}

extern "C" void kernel_launch(void* const* d_in, const int* in_sizes, int n_in,
                              void* d_out, int out_size, void* d_ws, size_t ws_size,
                              hipStream_t stream) {
  (void)in_sizes; (void)n_in; (void)d_ws; (void)ws_size; (void)out_size;
  const float* x     = (const float*)d_in[0];
  const float* tdist = (const float*)d_in[1];
  const float* W_ih  = (const float*)d_in[2];
  const float* W_hh  = (const float*)d_in[3];
  const float* bias  = (const float*)d_in[4];
  const float* W_d   = (const float*)d_in[5];
  const float* b_d   = (const float*)d_in[6];
  const float* fc2w  = (const float*)d_in[7];
  const float* fc2b  = (const float*)d_in[8];
  const float* basep = (const float*)d_in[9];
  float* out = (float*)d_out;

  dim3 grid(NBATCH / WAVES);   // 8192 blocks x 2 waves = 16384 batch elements
  dim3 block(THREADS);
  ctlstm_wmma_kernel<<<grid, block, 0, stream>>>(x, tdist, W_ih, W_hh, bias, W_d,
                                                 b_d, fc2w, fc2b, basep, out);
}
